// TurboQuantAttention_53171695125155
// MI455X (gfx1250) — matrix operand order repaired and checked by reading, not compiled, not run on hardware
//
#include <hip/hip_runtime.h>
#include <stdint.h>

// ---------------------------------------------------------------------------
// TurboQuantAttention for gfx1250 (MI455X, CDNA5, wave32).
// FA2-style: one 16-row Q tile per wave, 8 waves (256 thr) per block.
// Pipeline per 32-key stage:
//   1) global_load_async_to_lds_b128 stages raw f32 polar/qjl codes (ASYNCcnt,
//      double buffered),
//   2) block-cooperative dequant pass: each element dequantized ONCE into f16
//      LDS tiles (K row-major, V transposed),
//   3) waves run v_wmma_f32_16x16x32_f16:  S^T = K x Q^T (so the f32 C-layout
//      of the score tiles IS the f16 A-layout of P -> transpose-free P x V).
// Softmax uses raw v_exp_f32 / v_rcp_f32 (no libm denorm fixup sequences).
// ---------------------------------------------------------------------------

typedef __attribute__((ext_vector_type(16))) _Float16 v16h;
typedef __attribute__((ext_vector_type(8)))  _Float16 v8h;
typedef __attribute__((ext_vector_type(8)))  float    v8f;

#define POLAR_SCALE 0.125f
#define QJL_SCALE   0.01f

constexpr int S_LEN = 2048;   // sequence length
constexpr int DH    = 64;     // head dim
constexpr int BH    = 32;     // B*H
constexpr int KT    = 32;     // keys per pipeline stage
constexpr int LSTR  = 66;     // padded f32 per raw LDS row
constexpr int STG_F = KT * LSTR;        // raw floats per tensor per buffer
constexpr uint32_t TBYT = STG_F * 4;    // raw bytes per tensor per buffer
constexpr int NSTAGE = S_LEN / KT;      // 64
constexpr int KHSTR = 72;     // halfs per row of f16 K tile   (144 B rows)
constexpr int VTSTR = 40;     // halfs per row of f16 V^T tile (80 B rows)

// Raw hardware transcendentals: single v_exp_f32 / v_rcp_f32.
__device__ __forceinline__ float fast_exp2(float x) {
  return __builtin_amdgcn_exp2f(x);
}
__device__ __forceinline__ float fast_rcp(float x) {
  return __builtin_amdgcn_rcpf(x);
}

// ---- CDNA5 async copy: global -> LDS, 16 B per lane, tracked by ASYNCcnt ---
__device__ __forceinline__ void async_b128(uint32_t lds_byte, const float* g) {
  asm volatile("global_load_async_to_lds_b128 %0, %1, off"
               :: "v"(lds_byte), "v"((uint64_t)(uintptr_t)g)
               : "memory");
}
__device__ __forceinline__ void wait_async_le8() {
  asm volatile("s_wait_asynccnt 0x8" ::: "memory");
}
__device__ __forceinline__ void wait_async_0() {
  asm volatile("s_wait_asynccnt 0x0" ::: "memory");
}

// Block-cooperative raw-code stage copy: 32 rows x 64 f32 for each of the 4
// code tensors.  256 threads x 2 passes x 4 tensors = 8 async b128 per thread
// (per-wave ASYNCcnt +8 per stage).
__device__ __forceinline__ void stage_copy(uint32_t smbase, int buf, int kbase,
                                           const float* KP, const float* KQ,
                                           const float* VP, const float* VQ,
                                           int tid) {
  const int chunk = (tid & 15) * 4;
  const int row0  = tid >> 4;
#pragma unroll
  for (int pass = 0; pass < 2; ++pass) {
    const int row  = row0 + pass * 16;
    const int gidx = (kbase + row) * DH + chunk;
    const uint32_t lb = smbase + (uint32_t)((row * LSTR + chunk) * 4);
    async_b128(lb + (0u * 2u + (uint32_t)buf) * TBYT, KP + gidx);
    async_b128(lb + (1u * 2u + (uint32_t)buf) * TBYT, KQ + gidx);
    async_b128(lb + (2u * 2u + (uint32_t)buf) * TBYT, VP + gidx);
    async_b128(lb + (3u * 2u + (uint32_t)buf) * TBYT, VQ + gidx);
  }
}

__global__ __launch_bounds__(256)
void TurboQuantAttention_53171695125155_kernel(const float* __restrict__ q,
                                               const float* __restrict__ kp,
                                               const float* __restrict__ kq,
                                               const float* __restrict__ vp,
                                               const float* __restrict__ vq,
                                               float* __restrict__ out) {
  __shared__ __align__(16) float    rawsm[4 * 2 * STG_F]; // raw f32 codes
  __shared__ __align__(16) _Float16 Kh[KT * KHSTR];       // f16 K tile
  __shared__ __align__(16) _Float16 Vt[DH * VTSTR];       // f16 V^T tile

  const int tid  = threadIdx.x;
  const int lane = tid & 31;
  const int wv   = tid >> 5;          // wave 0..7
  const int ln   = lane & 15;
  const int hi   = lane >> 4;         // 0: lanes 0-15, 1: lanes 16-31
  const int bh   = blockIdx.y;
  const int qbase = blockIdx.x * 128 + wv * 16;

  const size_t hoff = (size_t)bh * S_LEN * DH;
  const float* Q  = q  + hoff;
  const float* KP = kp + hoff;
  const float* KQ = kq + hoff;
  const float* VP = vp + hoff;
  const float* VQ = vq + hoff;
  float*       O  = out + hoff;

  // Dequant-pass work assignment (each element exactly once per stage).
  const int krow = tid >> 3;          // 0..31
  const int kcol = (tid & 7) * 8;     // 0..56
  const int vd   = tid >> 2;          // 0..63
  const int vt0  = (tid & 3) * 8;     // 0..24

  // ---- Q tile once, pre-scaled by sm_scale*log2(e), in B-layout of Q^T.
  const float qscale = 0.125f * 1.44269504088896340736f;
  v16h bq[2];
#pragma unroll
  for (int c = 0; c < 2; ++c) {
    const float* src = Q + (size_t)(qbase + ln) * DH + c * 32 + hi * 16;
#pragma unroll
    for (int j = 0; j < 16; ++j) bq[c][j] = (_Float16)(src[j] * qscale);
  }

  v8f acc[4] = {};
  float m_run = -3.0e38f;
  float l_run = 0.0f;

  const uint32_t smbase = (uint32_t)(uintptr_t)rawsm;
  stage_copy(smbase, 0, 0, KP, KQ, VP, VQ, tid);

  for (int s = 0; s < NSTAGE; ++s) {
    const int buf = s & 1;
    if (s + 1 < NSTAGE) {
      stage_copy(smbase, buf ^ 1, (s + 1) * KT, KP, KQ, VP, VQ, tid);
      wait_async_le8();
    } else {
      wait_async_0();
    }
    __syncthreads();                       // raw stage s resident

    const float* kpB = rawsm + (0 * 2 + buf) * STG_F;
    const float* kqB = rawsm + (1 * 2 + buf) * STG_F;
    const float* vpB = rawsm + (2 * 2 + buf) * STG_F;
    const float* vqB = rawsm + (3 * 2 + buf) * STG_F;

    // ---- Cooperative dequant: raw codes -> f16 tiles (each element once).
    {
      const float* rp = kpB + krow * LSTR + kcol;
      const float* rq = kqB + krow * LSTR + kcol;
      v8h hk;
#pragma unroll
      for (int j = 0; j < 8; ++j)
        hk[j] = (_Float16)(rp[j] * POLAR_SCALE + rq[j] * QJL_SCALE);
      *(v8h*)(Kh + krow * KHSTR + kcol) = hk;

      v8h hv;
#pragma unroll
      for (int j = 0; j < 8; ++j) {
        const int t = vt0 + j;
        hv[j] = (_Float16)(vpB[t * LSTR + vd] * POLAR_SCALE +
                           vqB[t * LSTR + vd] * QJL_SCALE);
      }
      *(v8h*)(Vt + vd * VTSTR + vt0) = hv;   // transpose-on-store
    }
    __syncthreads();                       // f16 tiles ready

    // ---- S^T tiles: st[t] = K[t*16..+15] x Q^T.
    v8f st[2];
#pragma unroll
    for (int t = 0; t < 2; ++t) {
      v8f c0 = {};
#pragma unroll
      for (int c = 0; c < 2; ++c) {
        const int key   = t * 16 + ln;
        const int dbase = c * 32 + hi * 16;
        const _Float16* kb = Kh + key * KHSTR + dbase;
        v8h lo  = *(const v8h*)(kb);
        v8h hi8 = *(const v8h*)(kb + 8);
        v16h a = __builtin_shufflevector(lo, hi8, 0, 1, 2, 3, 4, 5, 6, 7,
                                         8, 9, 10, 11, 12, 13, 14, 15);
        c0 = __builtin_amdgcn_wmma_f32_16x16x32_f16(false, a, false, bq[c],
                                                    (short)0, c0, false, false);
      }
      st[t] = c0;
    }

    // ---- Online softmax (per query column m = ln; partner lane = l^16).
    float cmax = -3.0e38f;
#pragma unroll
    for (int r = 0; r < 8; ++r) cmax = fmaxf(cmax, fmaxf(st[0][r], st[1][r]));
    cmax = fmaxf(cmax, __shfl_xor(cmax, 16, 32));
    const float mnew  = fmaxf(m_run, cmax);
    const float alpha = fast_exp2(m_run - mnew);
    m_run = mnew;

    v16h ap;                       // C-layout of S^T == A-layout of P
    float psum = 0.0f;
#pragma unroll
    for (int r = 0; r < 8; ++r) {
      float p = fast_exp2(st[0][r] - mnew);
      ap[r] = (_Float16)p;  psum += p;
    }
#pragma unroll
    for (int r = 0; r < 8; ++r) {
      float p = fast_exp2(st[1][r] - mnew);
      ap[8 + r] = (_Float16)p;  psum += p;
    }
    psum += __shfl_xor(psum, 16, 32);
    l_run = l_run * alpha + psum;

    float av[8];
#pragma unroll
    for (int r = 0; r < 8; ++r) av[r] = __shfl(alpha, r + hi * 8, 32);
#pragma unroll
    for (int dd = 0; dd < 4; ++dd)
#pragma unroll
      for (int r = 0; r < 8; ++r) acc[dd][r] *= av[r];

    // ---- O += P(16x32) x V(32x64); B-operand = contiguous rows of V^T.
#pragma unroll
    for (int dd = 0; dd < 4; ++dd) {
      const _Float16* vb = Vt + (dd * 16 + ln) * VTSTR + hi * 8;
      v8h lo  = *(const v8h*)(vb);
      v8h hi8 = *(const v8h*)(vb + 16);
      v16h bv = __builtin_shufflevector(lo, hi8, 0, 1, 2, 3, 4, 5, 6, 7,
                                        8, 9, 10, 11, 12, 13, 14, 15);
      acc[dd] = __builtin_amdgcn_wmma_f32_16x16x32_f16(false, ap, false, bv,
                                                       (short)0, acc[dd],
                                                       false, false);
    }
    __syncthreads();               // all waves done with f16 tiles / raw bufs
  }

  // ---- Normalize (raw v_rcp_f32) and store.
  float sv[8];
#pragma unroll
  for (int r = 0; r < 8; ++r) sv[r] = fast_rcp(__shfl(l_run, r + hi * 8, 32));
#pragma unroll
  for (int dd = 0; dd < 4; ++dd)
#pragma unroll
    for (int r = 0; r < 8; ++r) {
      const int row = qbase + r + hi * 8;
      O[(size_t)row * DH + dd * 16 + ln] = acc[dd][r] * sv[r];
    }
}

extern "C" void kernel_launch(void* const* d_in, const int* in_sizes, int n_in,
                              void* d_out, int out_size, void* d_ws, size_t ws_size,
                              hipStream_t stream) {
  (void)in_sizes; (void)n_in; (void)out_size; (void)d_ws; (void)ws_size;
  const float* q  = (const float*)d_in[0];
  const float* kp = (const float*)d_in[1];
  const float* kq = (const float*)d_in[2];
  const float* vp = (const float*)d_in[3];
  const float* vq = (const float*)d_in[4];
  float* out = (float*)d_out;

  dim3 grid(S_LEN / 128, BH);   // 16 x 32 blocks
  dim3 block(256);              // 8 wave32 waves
  TurboQuantAttention_53171695125155_kernel<<<grid, block, 0, stream>>>(
      q, kp, kq, vp, vq, out);
}
